// VJEPASelfAttention_77781857730999
// MI455X (gfx1250) — matrix operand order repaired and checked by reading, not compiled, not run on hardware
//
#include <hip/hip_runtime.h>

// ---------------------------------------------------------------------------
// VJEPA self-attention, MI455X (gfx1250): bf16 WMMA GEMMs + fused flash attn
// Async global->LDS staging (ASYNCcnt) in the GEMM, global_prefetch in the
// flash kernel, permlane16 softmax reductions.
// ---------------------------------------------------------------------------

typedef __attribute__((ext_vector_type(16))) __bf16 v16bf;
typedef __attribute__((ext_vector_type(8)))  __bf16 bf16x8;
typedef __attribute__((ext_vector_type(8)))  float  v8f;
typedef int v4i __attribute__((__vector_size__(16)));

union frag16 { v16bf v; bf16x8 h[2]; };

static constexpr int Bc  = 4;
static constexpr int Sc  = 1568;
static constexpr int Dc  = 1024;
static constexpr int Hc  = 16;
static constexpr int HDc = 64;
static constexpr int Mrows = Bc * Sc;        // 6272

// ---- async global->LDS (CDNA5 GLOBAL_LOAD_ASYNC_TO_LDS_B128) --------------
// Prototype (from clang diagnostics): (v4i addrspace(1)*, v4i addrspace(3)*,
// imm int offset, imm int cpol).
#if __has_builtin(__builtin_amdgcn_global_load_async_to_lds_b128)
#define HAVE_ASYNC_LDS 1
#define AS1(p) ((__attribute__((address_space(1))) v4i*)(unsigned long long)(p))
#define AS3(p) ((__attribute__((address_space(3))) v4i*)(unsigned)(unsigned long long)(p))
#else
#define HAVE_ASYNC_LDS 0
#endif

__device__ __forceinline__ void wait_async0() {
#if __has_builtin(__builtin_amdgcn_s_wait_asynccnt)
  __builtin_amdgcn_s_wait_asynccnt(0);
#else
  asm volatile("s_wait_asynccnt 0" ::: "memory");
#endif
}

__device__ __forceinline__ __bf16 f2bf(float f) {
  unsigned u = __float_as_uint(f);
  unsigned r = u + 0x7FFFu + ((u >> 16) & 1u);   // round-to-nearest-even
  unsigned short s = (unsigned short)(r >> 16);
  __bf16 b;
  __builtin_memcpy(&b, &s, 2);
  return b;
}

__device__ __forceinline__ v8f wmma_bf16(v16bf a, v16bf b, v8f c) {
  return __builtin_amdgcn_wmma_f32_16x16x32_bf16(
      /*neg_a=*/false, a, /*neg_b=*/false, b,
      /*c_mod=*/(short)0, c, /*reuse_a=*/false, /*reuse_b=*/false);
}

// ---- cross-lane xor-butterfly within 16-lane halves (VALU, no LDS) --------
#if __has_builtin(__builtin_amdgcn_permlane16)
__device__ __forceinline__ float plane16(float x, unsigned s0, unsigned s1) {
  unsigned i = __float_as_uint(x);
  i = __builtin_amdgcn_permlane16(i, i, s0, s1, false, false);
  return __uint_as_float(i);
}
__device__ __forceinline__ float red_max16(float x) {
  x = fmaxf(x, plane16(x, 0x67452301u, 0xEFCDAB89u));  // xor 1
  x = fmaxf(x, plane16(x, 0x54761032u, 0xDCFE98BAu));  // xor 2
  x = fmaxf(x, plane16(x, 0x32107654u, 0xBA98FEDCu));  // xor 4
  x = fmaxf(x, plane16(x, 0xFEDCBA98u, 0x76543210u));  // xor 8
  return x;
}
__device__ __forceinline__ float red_sum16(float x) {
  x += plane16(x, 0x67452301u, 0xEFCDAB89u);
  x += plane16(x, 0x54761032u, 0xDCFE98BAu);
  x += plane16(x, 0x32107654u, 0xBA98FEDCu);
  x += plane16(x, 0xFEDCBA98u, 0x76543210u);
  return x;
}
#else
__device__ __forceinline__ float red_max16(float x) {
  for (int off = 1; off < 16; off <<= 1) x = fmaxf(x, __shfl_xor(x, off, 16));
  return x;
}
__device__ __forceinline__ float red_sum16(float x) {
  for (int off = 1; off < 16; off <<= 1) x += __shfl_xor(x, off, 16);
  return x;
}
#endif

// A-fragment (16x32 bf16): lanes 0-15 row M=lane&15, K in
// {kb..kb+15} contiguous, kb = 16 for upper lane half.
__device__ __forceinline__ v16bf load_frag_a(const __bf16* base, int stride,
                                             int r0, int lane) {
  frag16 u;
  const int m  = r0 + (lane & 15);
  const int kb = (lane >> 4) * 16;
  const __bf16* p = base + m * stride;
  u.h[0] = *(const bf16x8*)(p + kb);
  u.h[1] = *(const bf16x8*)(p + kb + 8);
  return u.v;
}

// B-fragment (32x16 bf16): lane holds column n = n0 + (lane&15),
// 16 contiguous K starting at (lane>=16 ? 16 : 0).
__device__ __forceinline__ v16bf load_frag_b(const __bf16* base, int stride,
                                             int n0, int lane) {
  frag16 u;
  const int n  = n0 + (lane & 15);
  const int kb = (lane >> 4) * 16;
  const __bf16* p = base + n * stride + kb;
  u.h[0] = *(const bf16x8*)(p);
  u.h[1] = *(const bf16x8*)(p + 8);
  return u.v;
}

// ---------------------------------------------------------------------------
// fp32 -> bf16 pack, 8 elements / thread
// ---------------------------------------------------------------------------
__global__ void cvt_f32_bf16_kernel(const float* __restrict__ src,
                                    __bf16* __restrict__ dst, int n8) {
  int i = blockIdx.x * blockDim.x + threadIdx.x;
  if (i >= n8) return;
  const float4* s = (const float4*)src + (size_t)i * 2;
  float4 v0 = s[0];
  float4 v1 = s[1];
  bf16x8 o;
  o[0] = f2bf(v0.x); o[1] = f2bf(v0.y); o[2] = f2bf(v0.z); o[3] = f2bf(v0.w);
  o[4] = f2bf(v1.x); o[5] = f2bf(v1.y); o[6] = f2bf(v1.z); o[7] = f2bf(v1.w);
  ((bf16x8*)dst)[i] = o;
}

// ---------------------------------------------------------------------------
// GEMM: C[M,N] = (A[M,K](bf16) @ W[N,K]^T + bias[N]) * oscale
// Block tile 128x128, 8 waves (4x2), wave tile 32x64 -> 8 WMMA / K-step.
// Double-buffered LDS; tile k+1 is DMA'd (async->LDS) or staged via VGPRs
// while the WMMAs of tile k run.
// MODE: 0 = bf16 row-major, 1 = f32 row-major, 2 = bf16 per-head transposed
//       ([B,H,HD,S], used for V so flash P@V reads contiguous keys).
// ---------------------------------------------------------------------------
template <int MODE>
__global__ __launch_bounds__(256) void gemm_bf16_nt_kernel(
    const __bf16* __restrict__ A, const __bf16* __restrict__ W,
    const float* __restrict__ bias, void* __restrict__ Cout,
    int M, int N, int K, float oscale) {
  __shared__ __align__(16) __bf16 As[2][128][40];   // 2 x 10 KB
  __shared__ __align__(16) __bf16 Bs[2][128][40];

  const int tid  = threadIdx.x;
  const int lane = tid & 31;
  const int wid  = tid >> 5;
  const int m0 = blockIdx.x * 128;
  const int n0 = blockIdx.y * 128;
  const int rb = (wid & 3) * 32;     // wave rows   [rb, rb+32)
  const int cb = (wid >> 2) * 64;    // wave cols   [cb, cb+64)

  v8f acc[2][4] = {};

  const int lrow = tid >> 1;            // 0..127
  const int lcg  = (tid & 1) * 16;      // 0 or 16

  const __bf16* ap = A + (size_t)(m0 + lrow) * K + lcg;
  const __bf16* wp = W + (size_t)(n0 + lrow) * K + lcg;
  const int nk = K / 32;

#if HAVE_ASYNC_LDS
  // ---- async DMA staging: global -> LDS, tracked by ASYNCcnt --------------
  auto issue_tile = [&](int p, int kk) {
    __builtin_amdgcn_global_load_async_to_lds_b128(
        AS1(ap + kk),     AS3(&As[p][lrow][lcg]),     0, 0);
    __builtin_amdgcn_global_load_async_to_lds_b128(
        AS1(ap + kk + 8), AS3(&As[p][lrow][lcg + 8]), 0, 0);
    __builtin_amdgcn_global_load_async_to_lds_b128(
        AS1(wp + kk),     AS3(&Bs[p][lrow][lcg]),     0, 0);
    __builtin_amdgcn_global_load_async_to_lds_b128(
        AS1(wp + kk + 8), AS3(&Bs[p][lrow][lcg + 8]), 0, 0);
  };
  issue_tile(0, 0);
  wait_async0();
  __syncthreads();
  int p = 0;
  for (int kt = 0; kt < nk; ++kt) {
    const bool more = (kt + 1) < nk;
    if (more) issue_tile(1 - p, (kt + 1) * 32);   // DMA overlaps WMMAs below

    v16bf a0 = load_frag_a(&As[p][0][0], 40, rb, lane);
    v16bf a1 = load_frag_a(&As[p][0][0], 40, rb + 16, lane);
#pragma unroll
    for (int j = 0; j < 4; ++j) {
      v16bf bj = load_frag_b(&Bs[p][0][0], 40, cb + j * 16, lane);
      acc[0][j] = wmma_bf16(a0, bj, acc[0][j]);
      acc[1][j] = wmma_bf16(a1, bj, acc[1][j]);
    }

    wait_async0();
    __syncthreads();
    p ^= 1;
  }
#else
  // ---- fallback: VGPR staging -------------------------------------------
  bf16x8 ra0, ra1, rw0, rw1;
  auto gload = [&](int kk) {
    ra0 = *(const bf16x8*)(ap + kk);
    ra1 = *(const bf16x8*)(ap + kk + 8);
    rw0 = *(const bf16x8*)(wp + kk);
    rw1 = *(const bf16x8*)(wp + kk + 8);
  };
  auto sstore = [&](int p) {
    *(bf16x8*)&As[p][lrow][lcg]     = ra0;
    *(bf16x8*)&As[p][lrow][lcg + 8] = ra1;
    *(bf16x8*)&Bs[p][lrow][lcg]     = rw0;
    *(bf16x8*)&Bs[p][lrow][lcg + 8] = rw1;
  };
  gload(0);
  sstore(0);
  __syncthreads();
  int p = 0;
  for (int kt = 0; kt < nk; ++kt) {
    const bool more = (kt + 1) < nk;
    if (more) gload((kt + 1) * 32);

    v16bf a0 = load_frag_a(&As[p][0][0], 40, rb, lane);
    v16bf a1 = load_frag_a(&As[p][0][0], 40, rb + 16, lane);
#pragma unroll
    for (int j = 0; j < 4; ++j) {
      v16bf bj = load_frag_b(&Bs[p][0][0], 40, cb + j * 16, lane);
      acc[0][j] = wmma_bf16(a0, bj, acc[0][j]);
      acc[1][j] = wmma_bf16(a1, bj, acc[1][j]);
    }

    if (more) sstore(1 - p);
    __syncthreads();
    p ^= 1;
  }
#endif

  // Epilogue. C layout: vgpr r -> M = r + 8*(lane>=16), N = lane&15.
  const int half = lane >> 4;
  const int nloc = lane & 15;
#pragma unroll
  for (int j = 0; j < 4; ++j) {
    const int col = n0 + cb + j * 16 + nloc;
    const float bv = bias[col];
#pragma unroll
    for (int i = 0; i < 2; ++i) {
#pragma unroll
      for (int r = 0; r < 8; ++r) {
        const int row = m0 + rb + i * 16 + r + half * 8;
        const float val = (acc[i][j][r] + bv) * oscale;
        if (MODE == 1) {
          ((float*)Cout)[(size_t)row * N + col] = val;
        } else if (MODE == 0) {
          ((__bf16*)Cout)[(size_t)row * N + col] = f2bf(val);
        } else {  // per-head transposed: [B, H, HD, S]
          const int bb = row / Sc;
          const int ss = row - bb * Sc;
          const int hh = col >> 6;
          const int dd = col & 63;
          ((__bf16*)Cout)[(((size_t)bb * Hc + hh) * HDc + dd) * Sc + ss] =
              f2bf(val);
        }
      }
    }
  }
}

// ---------------------------------------------------------------------------
// Fused flash attention, barrier-free. Grid: (ceil(S/128), H, B), 8 waves.
// Each wave owns 16 query rows; Q fragments persist in registers.
// K fragments load straight from global K (row-major, d contiguous);
// V fragments load straight from global Vt ([B,H,HD,S], keys contiguous).
// K/V tiles are reused by 13 blocks per (b,h) -> served from WGP$/L2, and
// next-tile lines are prefetched (global_prefetch) while this tile computes.
// Scale 1/sqrt(64) is pre-folded into Q. 1568 = 49 * 32 key tiles.
// ---------------------------------------------------------------------------
__global__ __launch_bounds__(256) void flash_attn_kernel(
    const __bf16* __restrict__ Q, const __bf16* __restrict__ Kd,
    const __bf16* __restrict__ Vt, __bf16* __restrict__ Ctx) {
  __shared__ __align__(16) __bf16 Ps[8][16][40];  // per-wave P bounce (10 KB)

  const int tid  = threadIdx.x;
  const int lane = tid & 31;
  const int wid  = tid >> 5;
  const int h = blockIdx.y;
  const int b = blockIdx.z;
  const int qbase = blockIdx.x * 128 + wid * 16;

  const size_t headBase = ((size_t)b * Sc) * Dc + (size_t)h * HDc;
  const __bf16* vtHead = Vt + ((size_t)(b * Hc + h) * HDc) * Sc;

  // Persistent Q A-fragments (16 q-rows x 64 d), clamped at the tail tile.
  int qm = qbase + (lane & 15);
  if (qm >= Sc) qm = Sc - 1;
  const int kb8 = (lane >> 4) * 16;
  const __bf16* qp = Q + headBase + (size_t)qm * Dc;
  frag16 uq0, uq1;
  uq0.h[0] = *(const bf16x8*)(qp + kb8);
  uq0.h[1] = *(const bf16x8*)(qp + kb8 + 8);
  uq1.h[0] = *(const bf16x8*)(qp + kb8 + 32);
  uq1.h[1] = *(const bf16x8*)(qp + kb8 + 40);

  float mrow[8], lrow[8];
  v8f o0 = {}, o1 = {}, o2 = {}, o3 = {};
#pragma unroll
  for (int r = 0; r < 8; ++r) { mrow[r] = -3.0e38f; lrow[r] = 0.0f; }

  const int half = lane >> 4;
  const int nloc = lane & 15;
  const int kseg = half * 16;     // which 16-K block this lane half supplies

  for (int kt = 0; kt < Sc / 32; ++kt) {
    const int kbase = kt * 32;

    // Prefetch next key tile's K and V lines into cache (global_prefetch).
    if (kt + 1 < Sc / 32) {
      __builtin_prefetch(Kd + headBase + (size_t)(kbase + 32 + nloc) * Dc + kseg, 0, 1);
      __builtin_prefetch(vtHead + (size_t)nloc * Sc + kbase + 32 + kseg, 0, 1);
    }

    // ---- scores: 16 q x 32 keys, two C tiles, d reduced over 64 ----------
    v8f sc0 = {}, sc1 = {};
    {
      // B-frags from K: column n = key, K-dim = d (contiguous in memory).
      const __bf16* kp0 = Kd + headBase + (size_t)(kbase + nloc) * Dc + kseg;
      const __bf16* kp1 = kp0 + 16 * Dc;  // keys +16
      frag16 k00, k01, k10, k11;
      k00.h[0] = *(const bf16x8*)(kp0);        // d 0..31
      k00.h[1] = *(const bf16x8*)(kp0 + 8);
      k01.h[0] = *(const bf16x8*)(kp0 + 32);   // d 32..63
      k01.h[1] = *(const bf16x8*)(kp0 + 40);
      k10.h[0] = *(const bf16x8*)(kp1);
      k10.h[1] = *(const bf16x8*)(kp1 + 8);
      k11.h[0] = *(const bf16x8*)(kp1 + 32);
      k11.h[1] = *(const bf16x8*)(kp1 + 40);
      sc0 = wmma_bf16(uq0.v, k00.v, sc0);
      sc0 = wmma_bf16(uq1.v, k01.v, sc0);
      sc1 = wmma_bf16(uq0.v, k10.v, sc1);
      sc1 = wmma_bf16(uq1.v, k11.v, sc1);
    }

    // ---- online softmax over this 32-key tile (VALU permlane reduce) -----
    float alpha[8];
#pragma unroll
    for (int r = 0; r < 8; ++r) {
      const float x0 = sc0[r];
      const float x1 = sc1[r];
      const float ml = red_max16(fmaxf(x0, x1));
      const float mn = fmaxf(mrow[r], ml);
      const float al = __expf(mrow[r] - mn);
      const float p0 = __expf(x0 - mn);
      const float p1 = __expf(x1 - mn);
      const float rs = red_sum16(p0 + p1);
      lrow[r] = lrow[r] * al + rs;
      mrow[r] = mn;
      alpha[r] = al;
      const int prow = r + half * 8;
      Ps[wid][prow][nloc]      = f2bf(p0);
      Ps[wid][prow][nloc + 16] = f2bf(p1);
    }
#pragma unroll
    for (int r = 0; r < 8; ++r) {
      o0[r] *= alpha[r]; o1[r] *= alpha[r];
      o2[r] *= alpha[r]; o3[r] *= alpha[r];
    }

    // ---- ctx += P(16x32) @ V(32x64) --------------------------------------
    v16bf pa = load_frag_a(&Ps[wid][0][0], 40, 0, lane);
    {
      // B-frags from Vt: column n = d, K-dim = key (contiguous in memory).
      const __bf16* vp = vtHead + (size_t)nloc * Sc + kbase + kseg;
      frag16 v0, v1, v2, v3;
      v0.h[0] = *(const bf16x8*)(vp);
      v0.h[1] = *(const bf16x8*)(vp + 8);
      v1.h[0] = *(const bf16x8*)(vp + 16 * Sc);
      v1.h[1] = *(const bf16x8*)(vp + 16 * Sc + 8);
      v2.h[0] = *(const bf16x8*)(vp + 32 * Sc);
      v2.h[1] = *(const bf16x8*)(vp + 32 * Sc + 8);
      v3.h[0] = *(const bf16x8*)(vp + 48 * Sc);
      v3.h[1] = *(const bf16x8*)(vp + 48 * Sc + 8);
      o0 = wmma_bf16(pa, v0.v, o0);
      o1 = wmma_bf16(pa, v1.v, o1);
      o2 = wmma_bf16(pa, v2.v, o2);
      o3 = wmma_bf16(pa, v3.v, o3);
    }
  }

  // Normalize and write ctx (bf16).
#pragma unroll
  for (int r = 0; r < 8; ++r) {
    const float inv = 1.0f / lrow[r];
    const int qr = qbase + r + half * 8;
    if (qr < Sc) {
      __bf16* cp = Ctx + ((size_t)b * Sc + qr) * Dc + (size_t)h * HDc + nloc;
      cp[0]  = f2bf(o0[r] * inv);
      cp[16] = f2bf(o1[r] * inv);
      cp[32] = f2bf(o2[r] * inv);
      cp[48] = f2bf(o3[r] * inv);
    }
  }
}

// ---------------------------------------------------------------------------
// Host-side orchestration
// ---------------------------------------------------------------------------
extern "C" void kernel_launch(void* const* d_in, const int* in_sizes, int n_in,
                              void* d_out, int out_size, void* d_ws,
                              size_t ws_size, hipStream_t stream) {
  const float* x  = (const float*)d_in[0];
  const float* Wq = (const float*)d_in[1];
  const float* bq = (const float*)d_in[2];
  const float* Wk = (const float*)d_in[3];
  const float* bk = (const float*)d_in[4];
  const float* Wv = (const float*)d_in[5];
  const float* bv = (const float*)d_in[6];
  const float* Wp = (const float*)d_in[7];
  const float* bp = (const float*)d_in[8];

  const size_t MD = (size_t)Mrows * Dc;   // 6,422,528
  const size_t DD = (size_t)Dc * Dc;      // 1,048,576

  __bf16* ws  = (__bf16*)d_ws;
  __bf16* xb  = ws;             // x bf16, later reused for ctx
  __bf16* wqb = xb  + MD;
  __bf16* wkb = wqb + DD;
  __bf16* wvb = wkb + DD;
  __bf16* wpb = wvb + DD;
  __bf16* qb  = wpb + DD;
  __bf16* kbf = qb  + MD;
  __bf16* vtb = kbf + MD;       // V stored per-head transposed [B,H,HD,S]
  __bf16* cxb = xb;             // reuse x buffer for ctx

  auto cvt = [&](const float* s, __bf16* d, size_t n) {
    const int n8 = (int)(n / 8);
    cvt_f32_bf16_kernel<<<(n8 + 255) / 256, 256, 0, stream>>>(s, d, n8);
  };
  cvt(x,  xb,  MD);
  cvt(Wq, wqb, DD);
  cvt(Wk, wkb, DD);
  cvt(Wv, wvb, DD);
  cvt(Wp, wpb, DD);

  dim3 gg(Mrows / 128, Dc / 128);  // 49 x 8
  // Q projection with softmax scale 1/sqrt(64) folded in.
  gemm_bf16_nt_kernel<0><<<gg, 256, 0, stream>>>(xb, wqb, bq, qb,
                                                 Mrows, Dc, Dc, 0.125f);
  gemm_bf16_nt_kernel<0><<<gg, 256, 0, stream>>>(xb, wkb, bk, kbf,
                                                 Mrows, Dc, Dc, 1.0f);
  // V projection written per-head transposed for contiguous P@V fragments.
  gemm_bf16_nt_kernel<2><<<gg, 256, 0, stream>>>(xb, wvb, bv, vtb,
                                                 Mrows, Dc, Dc, 1.0f);

  dim3 ga((Sc + 127) / 128, Hc, Bc);  // 13 x 16 x 4
  flash_attn_kernel<<<ga, 256, 0, stream>>>(qb, kbf, vtb, cxb);

  gemm_bf16_nt_kernel<1><<<gg, 256, 0, stream>>>(cxb, wpb, bp, d_out,
                                                 Mrows, Dc, Dc, 1.0f);
}
